// SO3Classifier_10101763080816
// MI455X (gfx1250) — matrix operand order repaired and checked by reading, not compiled, not run on hardware
//
#include <hip/hip_runtime.h>
#include <math.h>

// ---------------------------------------------------------------------------
// CDNA5 / gfx1250 types
// ---------------------------------------------------------------------------
typedef __attribute__((ext_vector_type(16))) __bf16 v16bf;
typedef __attribute__((ext_vector_type(8)))  float  v8f;
typedef __attribute__((ext_vector_type(4)))  unsigned int v4u;   // builtin vector: AS-cast friendly

#define DEV static __device__ __forceinline__

struct Q2 { v4u a, b; };   // 32-byte container for one v16bf fragment

// force global (addrspace 1) accesses so the GEMM body never emits flat_*
#define GLOAD4(p)       (*(const __attribute__((address_space(1))) v4u*)(unsigned long long)(p))
#define GSTORE_F(p, v)  (*(__attribute__((address_space(1))) float*)(unsigned long long)(p) = (v))
#define GSTORE_U16(p,v) (*(__attribute__((address_space(1))) unsigned short*)(unsigned long long)(p) = (v))

DEV int d_imin(int a, int b) { return a < b ? a : b; }
DEV int d_imax(int a, int b) { return a > b ? a : b; }

DEV unsigned short f2bf(float f) {
  unsigned int u = __builtin_bit_cast(unsigned int, f);
  u += 0x7FFFu + ((u >> 16) & 1u);          // round-to-nearest-even
  return (unsigned short)(u >> 16);
}

// ---------------------------------------------------------------------------
// Wigner-d constant generation (fp64, mirrors the numpy reference exactly)
// ---------------------------------------------------------------------------
DEV double factd(int n) { double r = 1.0; for (int i = 2; i <= n; ++i) r *= (double)i; return r; }
DEV double dipow(double x, int p) { double r = 1.0; for (int i = 0; i < p; ++i) r *= x; return r; }

DEV double small_d(int l, int mp, int m, double cb, double sb) {
  double pref = sqrt(factd(l + mp) * factd(l - mp) * factd(l + m) * factd(l - m));
  int k0 = d_imax(0, m - mp), k1 = d_imin(l + m, l - mp);
  double v = 0.0;
  for (int k = k0; k <= k1; ++k) {
    double den = factd(l + m - k) * factd(k) * factd(mp - m + k) * factd(l - mp - k);
    double t = (pref / den) * dipow(cb, 2 * l + m - mp - 2 * k) * dipow(sb, mp - m + 2 * k);
    v += ((mp - m + k) & 1) ? -t : t;
  }
  return v;
}

struct URow { int i0, i1, cnt; double re0, im0, re1, im1; };
DEV URow urow(int l, int a) {
  URow r; const double s2 = 0.70710678118654752440;
  int ma = a - l;
  if (ma == 0)      { r.cnt = 1; r.i0 = l;      r.re0 = 1.0; r.im0 = 0.0; r.i1 = 0;      r.re1 = 0.0;    r.im1 = 0.0; }
  else if (ma > 0)  { int m = ma;  double s = (m & 1) ? -1.0 : 1.0;
                      r.cnt = 2; r.i0 = l - m;  r.re0 = s2;  r.im0 = 0.0; r.i1 = l + m;  r.re1 = s * s2; r.im1 = 0.0; }
  else              { int m = -ma; double s = (m & 1) ? -1.0 : 1.0;
                      r.cnt = 2; r.i0 = l - m;  r.re0 = 0.0; r.im0 = s2;  r.i1 = l + m;  r.re1 = 0.0;    r.im1 = -s * s2; }
  return r;
}

// real-basis d: Re( U[a,:] d U^H[:,b] );  U rows have <=2 nonzeros
DEV double dreal_elem(int l, int a, int b, double cb, double sb) {
  URow ra = urow(l, a), rb = urow(l, b);
  int    ia[2] = { ra.i0, ra.i1 };  double rea[2] = { ra.re0, ra.re1 }, ima[2] = { ra.im0, ra.im1 };
  int    ib[2] = { rb.i0, rb.i1 };  double reb[2] = { rb.re0, rb.re1 }, imb[2] = { rb.im0, rb.im1 };
  double v = 0.0;
  for (int p = 0; p < ra.cnt; ++p)
    for (int q = 0; q < rb.cnt; ++q) {
      double w = rea[p] * reb[q] + ima[p] * imb[q];
      if (w != 0.0) v += w * small_d(l, ia[p] - l, ib[q] - l, cb, sb);
    }
  return v;
}

DEV int totc(int lmax) { int s = 0; for (int l = 0; l <= lmax; ++l) { int M = 2 * l + 1; s += M * M; } return s; }

// ---------------------------------------------------------------------------
// angle tables (one small kernel fills all grids/point-sets as doubles)
//  0: a1[20] 20: b1[20] 40: g1[20]      (so3_grid 20)
// 60: a2[10] 70: b2[10] 80: g2[10]      (so3_grid 10)
// 90: a3[6]  96: b3[6] 102: g3[6]       (so3_grid 6)
// 108: s2a[60] 168: s2b[60]             (s2 grid 60)
// 228: p5a[24] 252: p5b[24]             (s2 near-identity points)
// 276: p6a[168] 444: p6b[168] 612: p6g[168]  (so3 near-identity points)
// ---------------------------------------------------------------------------
__global__ void so3_fill_angles(double* g) {
  int i = threadIdx.x;
  const double PI = 3.14159265358979323846;
  if (i < 20) { g[  0 + i] = 2.0 * PI * i / 20.0; g[ 20 + i] = (i + 0.5) * PI / 20.0; g[ 40 + i] = 2.0 * PI * i / 20.0; }
  if (i < 10) { g[ 60 + i] = 2.0 * PI * i / 10.0; g[ 70 + i] = (i + 0.5) * PI / 10.0; g[ 80 + i] = 2.0 * PI * i / 10.0; }
  if (i <  6) { g[ 90 + i] = 2.0 * PI * i /  6.0; g[ 96 + i] = (i + 0.5) * PI /  6.0; g[102 + i] = 2.0 * PI * i /  6.0; }
  if (i < 60) { g[108 + i] = 2.0 * PI * i / 60.0; g[168 + i] = (i + 0.5) * PI / 60.0; }
  if (i < 24) { int a = i / 3, b = i % 3; g[228 + i] = 2.0 * PI * a / 8.0; g[252 + i] = (b + 1) * PI / 24.0; }
  if (i < 168) {
    int a = i / 24, b = (i / 8) % 3, k = i % 8;
    double al = 2.0 * PI * a / 7.0;
    double pg = -2.0 * PI + k * (4.0 * PI / 7.0);
    g[276 + i] = al; g[444 + i] = (b + 1) * PI / 24.0; g[612 + i] = pg - al;
  }
}

__global__ void so3_zero16(unsigned short* p, size_t n) {
  for (size_t i = blockIdx.x * (size_t)blockDim.x + threadIdx.x; i < n; i += (size_t)gridDim.x * blockDim.x) p[i] = 0;
}
__global__ void so3_zero32(float* p, size_t n) {
  for (size_t i = blockIdx.x * (size_t)blockDim.x + threadIdx.x; i < n; i += (size_t)gridDim.x * blockDim.x) p[i] = 0.0f;
}

// real Wigner-d tables:  out[ibeta*TOT + blockoff(l) + u*(2l+1) + v]
__global__ void so3_dr(int lmax, const double* betas, double* out, size_t total) {
  int TOT = totc(lmax);
  for (size_t idx = blockIdx.x * (size_t)blockDim.x + threadIdx.x; idx < total; idx += (size_t)gridDim.x * blockDim.x) {
    int ibeta = (int)(idx / TOT);
    int t = (int)(idx % TOT);
    int l = 0, off = 0;
    for (;;) { int M = 2 * l + 1; if (t < off + M * M) break; off += M * M; ++l; }
    int rem = t - off, M = 2 * l + 1, u = rem / M, v = rem % M;
    double be = betas[ibeta], cb = cos(0.5 * be), sb = sin(0.5 * be);
    out[idx] = dreal_elem(l, u, v, cb, sb);
  }
}

// flat Wigner-D matrices from dr tables. grid mode: n=(ia,ib,ig); points: ia=ib=ig=n.
__global__ void so3_flat(int lmax, int pointsMode, int na, int nb, int ng,
                         const double* A, const double* Bt, const double* Gm,
                         const double* dr, int drStride, double qwScale, int transposeOut,
                         unsigned short* out, int ldOut, size_t total) {
  int TOT = totc(lmax);
  for (size_t idx = blockIdx.x * (size_t)blockDim.x + threadIdx.x; idx < total; idx += (size_t)gridDim.x * blockDim.x) {
    int n = (int)(idx / TOT);
    int c = (int)(idx % TOT);
    int l = 0, off = 0;
    for (;;) { int M = 2 * l + 1; if (c < off + M * M) break; off += M * M; ++l; }
    int rem = c - off, M = 2 * l + 1, u = rem / M, v = rem % M;
    int ia, ib, ig;
    if (pointsMode) { ia = ib = ig = n; }
    else { ia = n / (nb * ng); ib = (n / ng) % nb; ig = n % ng; }
    int mu = u - l, mv = v - l;
    const double* row = dr + (size_t)ib * drStride + off;
    double d1 = row[u * M + v],           d2 = row[u * M + (M - 1 - v)];
    double d3 = row[(M - 1 - u) * M + v], d4 = row[(M - 1 - u) * M + (M - 1 - v)];
    double ga = Gm[ig] * mv, al = A[ia] * mu;
    double T1 = cos(ga) * d1 - sin(ga) * d2;
    double T2 = cos(ga) * d3 - sin(ga) * d4;
    double val = (cos(al) * T1 + sin(al) * T2) * sqrt((double)(2 * l + 1));
    if (qwScale != 0.0) val *= qwScale * sin(Bt[ib]);
    size_t o = transposeOut ? (size_t)c * ldOut + n : (size_t)n * ldOut + c;
    out[o] = f2bf((float)val);
  }
}

// spherical harmonics columns (center Wigner column), fp32 out.
__global__ void so3_sph(int lmax, int pointsMode, int na,
                        const double* A, const double* Bt,
                        const double* dr, int drStride, double qwScale,
                        float* out, int cols, size_t total) {
  for (size_t idx = blockIdx.x * (size_t)blockDim.x + threadIdx.x; idx < total; idx += (size_t)gridDim.x * blockDim.x) {
    int r = (int)(idx / cols);
    int c = (int)(idx % cols);
    int l = 0; while ((l + 1) * (l + 1) <= c) ++l;
    int u = c - l * l, M = 2 * l + 1, mu = u - l;
    int ia, ib;
    if (pointsMode) { ia = ib = r; } else { ib = r / na; ia = r % na; }
    int off = 0; for (int j = 0; j < l; ++j) { int Mj = 2 * j + 1; off += Mj * Mj; }
    const double* row = dr + (size_t)ib * drStride + off;
    double dc = row[u * M + l], dcr = row[(M - 1 - u) * M + l];
    double al = A[ia] * mu;
    double val = sqrt((double)(2 * l + 1)) * (cos(al) * dc + sin(al) * dcr);
    if (qwScale != 0.0) val *= qwScale * sin(Bt[ib]);
    out[idx] = (float)val;
  }
}

// ---------------------------------------------------------------------------
// WMMA bf16 NT GEMM:  C[M,Npad] = scale * A[M,K] * B'[Npad,K]^T  (+optional relu)
// wave32: one wave -> 32x64 C tile (2 A-frags x 4 B-frags -> 8 WMMA / k-step).
// 8 waves/block as 2M x 4N -> 64x256 block tile. Branch-free templated bodies;
// edge tiles dispatch to smaller instantiations with *scalar* (SALU) branches.
// All memory traffic forced to addrspace(1) => pure global_load/store (no flat,
// no DScnt coupling). K % 32 == 0, M % 16 == 0, Npad % 16 == 0.
// ---------------------------------------------------------------------------
template <int MT, int NT>
DEV void gemm_tile(const unsigned short* __restrict__ A, long lda,
                   const unsigned short* __restrict__ B, long ldb,
                   float* __restrict__ C, unsigned short* __restrict__ Cbf, long ldc,
                   int m0, int n0, int K, float scale, int relu, int lane) {
  int ln = lane & 15;
  int kaOff = (lane & 16) ? 8 : 0;      // A fragment K sub-base (ISA 16-bit A layout)
  int kbOff = (lane & 16) ? 8 : 0;      // B fragment K sub-base (ISA 16-bit B layout)

  const unsigned short* pa[MT];
  const unsigned short* pb[NT];
#pragma unroll
  for (int i = 0; i < MT; ++i) pa[i] = A + (size_t)(m0 + 16 * i + ln) * lda + kaOff;
#pragma unroll
  for (int j = 0; j < NT; ++j) pb[j] = B + (size_t)(n0 + 16 * j + ln) * ldb + kbOff;

  v8f acc[MT][NT] = {};
  for (int k = 0; k < K; k += 32) {
    v16bf a[MT];
#pragma unroll
    for (int i = 0; i < MT; ++i) {
      Q2 q; q.a = GLOAD4(pa[i]); q.b = GLOAD4(pa[i] + 16);
      a[i] = __builtin_bit_cast(v16bf, q);
      __builtin_prefetch(pa[i] + 256, 0, 3);   // 512B ahead on the A stream
      pa[i] += 32;
    }
#pragma unroll
    for (int j = 0; j < NT; ++j) {
      Q2 q; q.a = GLOAD4(pb[j]); q.b = GLOAD4(pb[j] + 16);
      v16bf b = __builtin_bit_cast(v16bf, q);
      __builtin_prefetch(pb[j] + 256, 0, 3);   // 512B ahead on the B stream
      pb[j] += 32;
#pragma unroll
      for (int i = 0; i < MT; ++i)
        acc[i][j] = __builtin_amdgcn_wmma_f32_16x16x32_bf16(false, a[i], false, b, (short)0, acc[i][j], false, false);
    }
  }

  int mb = (lane & 16) ? 8 : 0;
#pragma unroll
  for (int i = 0; i < MT; ++i)
#pragma unroll
    for (int j = 0; j < NT; ++j) {
      int nc = n0 + 16 * j + ln;
#pragma unroll
      for (int r = 0; r < 8; ++r) {
        float v = acc[i][j][r] * scale;
        if (relu) v = v > 0.0f ? v : 0.0f;
        size_t o = (size_t)(m0 + 16 * i + mb + r) * ldc + nc;
        GSTORE_F(C + o, v);
        if (Cbf) GSTORE_U16(Cbf + o, f2bf(v));
      }
    }
}

__global__ void __launch_bounds__(256)
so3_gemm_nt(const unsigned short* __restrict__ A, long lda,
            const unsigned short* __restrict__ B, long ldb,
            float* __restrict__ C, unsigned short* __restrict__ Cbf, long ldc,
            int M, int Npad, int K, float scale, int relu) {
  int lane = (int)(threadIdx.x & 31);
  int wv = __builtin_amdgcn_readfirstlane((int)(threadIdx.x >> 5));  // scalar wave id
  int wm = wv >> 2, wn = wv & 3;
  int m0 = (int)blockIdx.y * 64 + wm * 32;
  int n0 = (int)blockIdx.x * 256 + wn * 64;
  if (m0 >= M || n0 >= Npad) return;            // uniform (SALU) branch
  int mt = (M - m0) >= 32 ? 2 : 1;
  int nt = (Npad - n0) >> 4; if (nt > 4) nt = 4;
  if (mt == 2) {
    switch (nt) {
      case 4:  gemm_tile<2, 4>(A, lda, B, ldb, C, Cbf, ldc, m0, n0, K, scale, relu, lane); break;
      case 3:  gemm_tile<2, 3>(A, lda, B, ldb, C, Cbf, ldc, m0, n0, K, scale, relu, lane); break;
      case 2:  gemm_tile<2, 2>(A, lda, B, ldb, C, Cbf, ldc, m0, n0, K, scale, relu, lane); break;
      default: gemm_tile<2, 1>(A, lda, B, ldb, C, Cbf, ldc, m0, n0, K, scale, relu, lane); break;
    }
  } else {
    switch (nt) {
      case 4:  gemm_tile<1, 4>(A, lda, B, ldb, C, Cbf, ldc, m0, n0, K, scale, relu, lane); break;
      case 3:  gemm_tile<1, 3>(A, lda, B, ldb, C, Cbf, ldc, m0, n0, K, scale, relu, lane); break;
      case 2:  gemm_tile<1, 2>(A, lda, B, ldb, C, Cbf, ldc, m0, n0, K, scale, relu, lane); break;
      default: gemm_tile<1, 1>(A, lda, B, ldb, C, Cbf, ldc, m0, n0, K, scale, relu, lane); break;
    }
  }
}

// ---------------------------------------------------------------------------
// small fp32 kernels (FromS2Grid projection, psi1, FC head)
// ---------------------------------------------------------------------------
__global__ void so3_from_s2(const float* x, const float* FS2, float* coeff, size_t total) {
  for (size_t idx = blockIdx.x * (size_t)blockDim.x + threadIdx.x; idx < total; idx += (size_t)gridDim.x * blockDim.x) {
    int b = (int)(idx / 441), c = (int)(idx % 441);
    double s = 0.0;
    for (int ib = 0; ib < 60; ++ib)
      for (int ia = 0; ia < 60; ++ia)
        s += (double)x[(b * 60 + ia) * 60 + ib] * (double)FS2[(size_t)(ib * 60 + ia) * 441 + c];
    coeff[idx] = (float)s;
  }
}

__global__ void so3_psi1(const float* conv1, const float* YK, float* psi1, size_t total) {
  const float sc = 0.20412414523193150818f; // 1/sqrt(24)
  for (size_t idx = blockIdx.x * (size_t)blockDim.x + threadIdx.x; idx < total; idx += (size_t)gridDim.x * blockDim.x) {
    int o = (int)(idx / 441), c = (int)(idx % 441);
    float s = 0.0f;
    for (int n = 0; n < 24; ++n) s += conv1[o * 24 + n] * YK[n * 441 + c];
    psi1[idx] = s * sc;
  }
}

// s2_linear with fi=1: outer product coeff x psi1 -> bf16 h1 [512, 12352]
__global__ void so3_s2lin(const float* coeff, const float* psi1, unsigned short* h1, size_t total) {
  for (size_t idx = blockIdx.x * (size_t)blockDim.x + threadIdx.x; idx < total; idx += (size_t)gridDim.x * blockDim.x) {
    int r = (int)(idx / 12352), c = (int)(idx % 12352);
    if (c >= 12341) { h1[idx] = 0; continue; }
    int l = 0, off = 0;
    for (;;) { int M = 2 * l + 1; if (c < off + M * M) break; off += M * M; ++l; }
    int rem = c - off, M = 2 * l + 1, u = rem / M, m = rem % M;
    int b = r >> 6, o = r & 63;
    float v = coeff[b * 441 + l * l + m] * psi1[o * 441 + l * l + u];
    h1[idx] = f2bf(v);
  }
}

// f32 -> bf16 with column padding
__global__ void so3_cvt(const float* in, long ldin, unsigned short* out, long ldout,
                        int cols, int colsPad, size_t total) {
  for (size_t idx = blockIdx.x * (size_t)blockDim.x + threadIdx.x; idx < total; idx += (size_t)gridDim.x * blockDim.x) {
    int r = (int)(idx / colsPad), c = (int)(idx % colsPad);
    out[(size_t)r * ldout + c] = (c < cols) ? f2bf(in[(size_t)r * ldin + c]) : (unsigned short)0;
  }
}

// per-l permutes for _so3_linear  (A2[(b,m),(i,u)], W2T[(o,v),(i,u)], O -> h)
__global__ void so3_permA(const float* h, long ldh, int fi, int off, int d,
                          unsigned short* A2, size_t total) {
  int KD = fi * d;
  for (size_t idx = blockIdx.x * (size_t)blockDim.x + threadIdx.x; idx < total; idx += (size_t)gridDim.x * blockDim.x) {
    int r = (int)(idx / KD), c = (int)(idx % KD);
    if (r >= 8 * d) { A2[idx] = 0; continue; }
    int i = c / d, u = c % d, b = r / d, m = r % d;
    A2[idx] = f2bf(h[(size_t)(b * fi + i) * ldh + off + u * d + m]);
  }
}
__global__ void so3_permW(const float* psi, long ldpsi, int fi, int fo, int off, int d,
                          unsigned short* W2T, size_t total) {
  int KD = fi * d;
  for (size_t idx = blockIdx.x * (size_t)blockDim.x + threadIdx.x; idx < total; idx += (size_t)gridDim.x * blockDim.x) {
    int row = (int)(idx / KD), c = (int)(idx % KD);
    int o = row / d, v = row % d, i = c / d, u = c % d;
    W2T[idx] = f2bf(psi[(size_t)(i * fo + o) * ldpsi + off + u * d + v]);
  }
}
__global__ void so3_permO(const float* O, int fo, int off, int d, float* hout, long ldh, size_t total) {
  int dd = d * d;
  for (size_t idx = blockIdx.x * (size_t)blockDim.x + threadIdx.x; idx < total; idx += (size_t)gridDim.x * blockDim.x) {
    int b = (int)(idx / ((size_t)fo * dd));
    int rem = (int)(idx % ((size_t)fo * dd));
    int o = rem / dd, r2 = rem % dd, v = r2 / d, m = r2 % d;
    hout[(size_t)(b * fo + o) * ldh + off + v * d + m] = O[(size_t)(b * d + m) * (fo * d) + o * d + v];
  }
}

__global__ void so3_fc1(const float* h6, const float* w1, const float* b1, float* t, size_t total) {
  for (size_t idx = blockIdx.x * (size_t)blockDim.x + threadIdx.x; idx < total; idx += (size_t)gridDim.x * blockDim.x) {
    int b = (int)(idx / 128), j = (int)(idx % 128);
    float s = b1[j];
    for (int i = 0; i < 256; ++i) s += h6[(size_t)(b * 256 + i) * 16] * w1[i * 128 + j];
    t[idx] = s > 0.0f ? s : 0.0f;
  }
}
__global__ void so3_fc2(const float* t, const float* w2, const float* b2, float* out, size_t total) {
  for (size_t idx = blockIdx.x * (size_t)blockDim.x + threadIdx.x; idx < total; idx += (size_t)gridDim.x * blockDim.x) {
    int b = (int)(idx / 10), k = (int)(idx % 10);
    float s = b2[k];
    for (int j = 0; j < 128; ++j) s += t[b * 128 + j] * w2[j * 10 + k];
    out[idx] = s;
  }
}

// ---------------------------------------------------------------------------
// host orchestration
// ---------------------------------------------------------------------------
extern "C" void kernel_launch(void* const* d_in, const int* in_sizes, int n_in,
                              void* d_out, int out_size, void* d_ws, size_t ws_size,
                              hipStream_t stream) {
  (void)in_sizes; (void)n_in; (void)out_size;
  const float* x      = (const float*)d_in[0];
  const float* conv1w = (const float*)d_in[1];
  const float* conv2w = (const float*)d_in[2];
  const float* conv3w = (const float*)d_in[3];
  const float* fcw1   = (const float*)d_in[4];
  const float* fcb1   = (const float*)d_in[5];
  const float* fcw2   = (const float*)d_in[6];
  const float* fcb2   = (const float*)d_in[7];
  float* out = (float*)d_out;

  const double PI = 3.14159265358979323846;
  const int TOT1 = 12341, TOT2 = 1771, TOT3 = 286;

  char* base = (char*)d_ws; size_t off = 0;
  auto alloc = [&](size_t bytes) -> char* {
    char* p = base + off; off = (off + bytes + 255) & ~(size_t)255; return p;
  };

  double* ANG = (double*)alloc(780 * 8);
  double* T1  = (double*)alloc((size_t)20  * TOT1 * 8);
  double* T2  = (double*)alloc((size_t)10  * TOT2 * 8);
  double* T3  = (double*)alloc((size_t)6   * TOT3 * 8);
  double* T4  = (double*)alloc((size_t)60  * TOT1 * 8);
  double* T5  = (double*)alloc((size_t)24  * TOT1 * 8);
  double* T6  = (double*)alloc((size_t)168 * TOT2 * 8);

  unsigned short* D1IN   = (unsigned short*)alloc((size_t)8000 * 12352 * 2);
  unsigned short* D1BWDT = (unsigned short*)alloc((size_t)1792 * 8000 * 2);
  unsigned short* D2IN   = (unsigned short*)alloc((size_t)1024 * 1792 * 2);
  unsigned short* D2BWDT = (unsigned short*)alloc((size_t)288  * 1024 * 2);
  unsigned short* D3IN   = (unsigned short*)alloc((size_t)224  * 288  * 2);
  unsigned short* D3BWDT = (unsigned short*)alloc((size_t)16   * 224  * 2);
  unsigned short* DK2T   = (unsigned short*)alloc((size_t)1792 * 192  * 2);
  unsigned short* DK3T   = (unsigned short*)alloc((size_t)288  * 192  * 2);
  float* FS2  = (float*)alloc((size_t)3600 * 441 * 4);
  float* YK   = (float*)alloc((size_t)24   * 441 * 4);

  float* COEF = (float*)alloc((size_t)8  * 441 * 4);
  float* PSI1 = (float*)alloc((size_t)64 * 441 * 4);
  unsigned short* H1  = (unsigned short*)alloc((size_t)512 * 12352 * 2);
  float* F1           = (float*)alloc((size_t)512 * 8000 * 4);
  unsigned short* F1B = (unsigned short*)alloc((size_t)512 * 8000 * 2);
  float* H2           = (float*)alloc((size_t)512 * 1792 * 4);
  unsigned short* C2B = (unsigned short*)alloc((size_t)8192 * 192 * 2);
  float* PSI2         = (float*)alloc((size_t)8192 * 1792 * 4);
  unsigned short* A2  = (unsigned short*)alloc((size_t)176 * 1408 * 2);
  unsigned short* W2T = (unsigned short*)alloc((size_t)2816 * 1408 * 2);
  float* OBUF         = (float*)alloc((size_t)176 * 2816 * 4);
  float* H3           = (float*)alloc((size_t)1024 * 1792 * 4);
  unsigned short* H3B = (unsigned short*)alloc((size_t)1024 * 1792 * 2);
  float* F2           = (float*)alloc((size_t)1024 * 1024 * 4);
  unsigned short* F2B = (unsigned short*)alloc((size_t)1024 * 1024 * 2);
  float* H4           = (float*)alloc((size_t)1024 * 288 * 4);
  unsigned short* C3B = (unsigned short*)alloc((size_t)32768 * 192 * 2);
  float* PSI3         = (float*)alloc((size_t)32768 * 288 * 4);
  float* H5           = (float*)alloc((size_t)2048 * 288 * 4);
  unsigned short* H5B = (unsigned short*)alloc((size_t)2048 * 288 * 2);
  float* F3           = (float*)alloc((size_t)2048 * 224 * 4);
  unsigned short* F3B = (unsigned short*)alloc((size_t)2048 * 224 * 2);
  float* H6           = (float*)alloc((size_t)2048 * 16 * 4);
  float* TFC          = (float*)alloc((size_t)8 * 128 * 4);

  if (off > ws_size) return;  // workspace too small: do nothing (deterministic)

  auto nb = [](size_t n) -> unsigned {
    size_t b = (n + 255) / 256; if (b > 8192) b = 8192; return (unsigned)b;
  };
  auto gemm = [&](const unsigned short* A, long lda, const unsigned short* B, long ldb,
                  float* C, unsigned short* Cbf, long ldc, int M, int Npad, int K,
                  float scale, int relu) {
    dim3 g((Npad + 255) / 256, (M + 63) / 64);
    so3_gemm_nt<<<g, 256, 0, stream>>>(A, lda, B, ldb, C, Cbf, ldc, M, Npad, K, scale, relu);
  };

  // ---- constants ----
  so3_fill_angles<<<1, 256, 0, stream>>>(ANG);

  { size_t n;
    n = (size_t)8000 * 12352; so3_zero16<<<nb(n), 256, 0, stream>>>(D1IN, n);
    n = (size_t)1792 * 8000;  so3_zero16<<<nb(n), 256, 0, stream>>>(D1BWDT, n);
    n = (size_t)1024 * 1792;  so3_zero16<<<nb(n), 256, 0, stream>>>(D2IN, n);
    n = (size_t)288 * 1024;   so3_zero16<<<nb(n), 256, 0, stream>>>(D2BWDT, n);
    n = (size_t)224 * 288;    so3_zero16<<<nb(n), 256, 0, stream>>>(D3IN, n);
    n = (size_t)16 * 224;     so3_zero16<<<nb(n), 256, 0, stream>>>(D3BWDT, n);
    n = (size_t)1792 * 192;   so3_zero16<<<nb(n), 256, 0, stream>>>(DK2T, n);
    n = (size_t)288 * 192;    so3_zero16<<<nb(n), 256, 0, stream>>>(DK3T, n);
    n = (size_t)1024 * 1792;  so3_zero32<<<nb(n), 256, 0, stream>>>(H3, n);
    n = (size_t)2048 * 288;   so3_zero32<<<nb(n), 256, 0, stream>>>(H5, n);
  }

  { size_t n;
    n = (size_t)20 * TOT1;  so3_dr<<<nb(n), 256, 0, stream>>>(20, ANG + 20,  T1, n);
    n = (size_t)10 * TOT2;  so3_dr<<<nb(n), 256, 0, stream>>>(10, ANG + 70,  T2, n);
    n = (size_t)6 * TOT3;   so3_dr<<<nb(n), 256, 0, stream>>>(5,  ANG + 96,  T3, n);
    n = (size_t)60 * TOT1;  so3_dr<<<nb(n), 256, 0, stream>>>(20, ANG + 168, T4, n);
    n = (size_t)24 * TOT1;  so3_dr<<<nb(n), 256, 0, stream>>>(20, ANG + 252, T5, n);
    n = (size_t)168 * TOT2; so3_dr<<<nb(n), 256, 0, stream>>>(10, ANG + 444, T6, n);
  }

  { size_t n;
    n = (size_t)8000 * TOT1;
    so3_flat<<<nb(n), 256, 0, stream>>>(20, 0, 20, 20, 20, ANG + 0,  ANG + 20,  ANG + 40,  T1, TOT1, 0.0,          0, D1IN,   12352, n);
    n = (size_t)8000 * TOT2;
    so3_flat<<<nb(n), 256, 0, stream>>>(10, 0, 20, 20, 20, ANG + 0,  ANG + 20,  ANG + 40,  T1, TOT1, PI / 16000.0, 1, D1BWDT, 8000,  n);
    n = (size_t)1000 * TOT2;
    so3_flat<<<nb(n), 256, 0, stream>>>(10, 0, 10, 10, 10, ANG + 60, ANG + 70,  ANG + 80,  T2, TOT2, 0.0,          0, D2IN,   1792,  n);
    n = (size_t)1000 * TOT3;
    so3_flat<<<nb(n), 256, 0, stream>>>(5,  0, 10, 10, 10, ANG + 60, ANG + 70,  ANG + 80,  T2, TOT2, PI / 2000.0,  1, D2BWDT, 1024,  n);
    n = (size_t)216 * TOT3;
    so3_flat<<<nb(n), 256, 0, stream>>>(5,  0, 6, 6, 6,    ANG + 90, ANG + 96,  ANG + 102, T3, TOT3, 0.0,          0, D3IN,   288,   n);
    n = (size_t)216 * 1;
    so3_flat<<<nb(n), 256, 0, stream>>>(0,  0, 6, 6, 6,    ANG + 90, ANG + 96,  ANG + 102, T3, TOT3, PI / 432.0,   1, D3BWDT, 224,   n);
    n = (size_t)168 * TOT2;
    so3_flat<<<nb(n), 256, 0, stream>>>(10, 1, 168, 168, 168, ANG + 276, ANG + 444, ANG + 612, T6, TOT2, 0.0,      1, DK2T,   192,   n);
    n = (size_t)168 * TOT3;
    so3_flat<<<nb(n), 256, 0, stream>>>(5,  1, 168, 168, 168, ANG + 276, ANG + 444, ANG + 612, T6, TOT2, 0.0,      1, DK3T,   192,   n);
  }

  { size_t n;
    n = (size_t)3600 * 441;
    so3_sph<<<nb(n), 256, 0, stream>>>(20, 0, 60, ANG + 108, ANG + 168, T4, TOT1, PI / 7200.0, FS2, 441, n);
    n = (size_t)24 * 441;
    so3_sph<<<nb(n), 256, 0, stream>>>(20, 1, 24, ANG + 228, ANG + 252, T5, TOT1, 0.0,         YK,  441, n);
  }

  // ---- network ----
  { size_t n = (size_t)8 * 441;   so3_from_s2<<<nb(n), 256, 0, stream>>>(x, FS2, COEF, n); }
  { size_t n = (size_t)64 * 441;  so3_psi1<<<nb(n), 256, 0, stream>>>(conv1w, YK, PSI1, n); }
  { size_t n = (size_t)512 * 12352; so3_s2lin<<<nb(n), 256, 0, stream>>>(COEF, PSI1, H1, n); }

  // act1: f1 = relu(h1 * D1_IN^T);  h2 = f1 * D1_BWD
  gemm(H1, 12352, D1IN, 12352, F1, F1B, 8000, 512, 8000, 12352, 1.0f, 1);
  gemm(F1B, 8000, D1BWDT, 8000, H2, nullptr, 1792, 512, 1792, 8000, 1.0f, 0);

  // psi2 = conv2_w * D_K2 / sqrt(168)
  { size_t n = (size_t)8192 * 192; so3_cvt<<<nb(n), 256, 0, stream>>>(conv2w, 168, C2B, 192, 168, 192, n); }
  gemm(C2B, 192, DK2T, 192, PSI2, nullptr, 1792, 8192, 1792, 192, 0.07715167498104595f, 0);

  // so3_linear (64 -> 128) per l
  { int o2 = 0;
    for (int l = 0; l <= 10; ++l) {
      int d = 2 * l + 1;
      int R16 = ((8 * d + 15) / 16) * 16;
      int Kd = 64 * d, Nd = 128 * d;
      size_t n;
      n = (size_t)R16 * Kd;        so3_permA<<<nb(n), 256, 0, stream>>>(H2, 1792, 64, o2, d, A2, n);
      n = (size_t)Nd * Kd;         so3_permW<<<nb(n), 256, 0, stream>>>(PSI2, 1792, 64, 128, o2, d, W2T, n);
      gemm(A2, Kd, W2T, Kd, OBUF, nullptr, Nd, R16, Nd, Kd, 1.0f / sqrtf((float)(64 * d)), 0);
      n = (size_t)8 * 128 * d * d; so3_permO<<<nb(n), 256, 0, stream>>>(OBUF, 128, o2, d, H3, 1792, n);
      o2 += d * d;
    }
  }

  // act2
  { size_t n = (size_t)1024 * 1792; so3_cvt<<<nb(n), 256, 0, stream>>>(H3, 1792, H3B, 1792, 1792, 1792, n); }
  gemm(H3B, 1792, D2IN, 1792, F2, F2B, 1024, 1024, 1024, 1792, 1.0f, 1);
  gemm(F2B, 1024, D2BWDT, 1024, H4, nullptr, 288, 1024, 288, 1024, 1.0f, 0);

  // psi3 = conv3_w * D_K3 / sqrt(168)
  { size_t n = (size_t)32768 * 192; so3_cvt<<<nb(n), 256, 0, stream>>>(conv3w, 168, C3B, 192, 168, 192, n); }
  gemm(C3B, 192, DK3T, 192, PSI3, nullptr, 288, 32768, 288, 192, 0.07715167498104595f, 0);

  // so3_linear (128 -> 256) per l
  { int o3 = 0;
    for (int l = 0; l <= 5; ++l) {
      int d = 2 * l + 1;
      int R16 = ((8 * d + 15) / 16) * 16;
      int Kd = 128 * d, Nd = 256 * d;
      size_t n;
      n = (size_t)R16 * Kd;        so3_permA<<<nb(n), 256, 0, stream>>>(H4, 288, 128, o3, d, A2, n);
      n = (size_t)Nd * Kd;         so3_permW<<<nb(n), 256, 0, stream>>>(PSI3, 288, 128, 256, o3, d, W2T, n);
      gemm(A2, Kd, W2T, Kd, OBUF, nullptr, Nd, R16, Nd, Kd, 1.0f / sqrtf((float)(128 * d)), 0);
      n = (size_t)8 * 256 * d * d; so3_permO<<<nb(n), 256, 0, stream>>>(OBUF, 256, o3, d, H5, 288, n);
      o3 += d * d;
    }
  }

  // act3 (lmax_out = 0)
  { size_t n = (size_t)2048 * 288; so3_cvt<<<nb(n), 256, 0, stream>>>(H5, 288, H5B, 288, 288, 288, n); }
  gemm(H5B, 288, D3IN, 288, F3, F3B, 224, 2048, 224, 288, 1.0f, 1);
  gemm(F3B, 224, D3BWDT, 224, H6, nullptr, 16, 2048, 16, 224, 1.0f, 0);

  // FC head
  { size_t n = (size_t)8 * 128; so3_fc1<<<nb(n), 256, 0, stream>>>(H6, fcw1, fcb1, TFC, n); }
  { size_t n = (size_t)8 * 10;  so3_fc2<<<nb(n), 256, 0, stream>>>(TFC, fcw2, fcb2, out, n); }
}
